// Encoder_GCN_decoder_11596411699261
// MI455X (gfx1250) — matrix operand order repaired and checked by reading, not compiled, not run on hardware
//
#include <hip/hip_runtime.h>
#include <hip/hip_bf16.h>

typedef __attribute__((ext_vector_type(16))) _Float16 v16h;
typedef __attribute__((ext_vector_type(8)))  _Float16 v8h;
typedef __attribute__((ext_vector_type(4)))  _Float16 v4h;
typedef __attribute__((ext_vector_type(8)))  float    v8f;

#define GEMM_WAVES 4

__device__ __forceinline__ float gelu_exact(float v) {
  return 0.5f * v * (1.0f + erff(v * 0.70710678118654752440f));
}

__device__ __forceinline__ void atomAddF(float* a, float v) {
  __hip_atomic_fetch_add(a, v, __ATOMIC_RELAXED, __HIP_MEMORY_SCOPE_AGENT);
}

__device__ __forceinline__ void atomMaxF(float* addr, float v) {
  int* ai = (int*)addr;
  int old = __hip_atomic_load(ai, __ATOMIC_RELAXED, __HIP_MEMORY_SCOPE_AGENT);
  while (__int_as_float(old) < v) {
    int assumed = old;
    old = atomicCAS(ai, assumed, __float_as_int(v));
    if (old == assumed) break;
  }
}

__device__ __forceinline__ v16h cat16(v8h lo, v8h hi) {
  return __builtin_shufflevector(lo, hi, 0, 1, 2, 3, 4, 5, 6, 7,
                                         8, 9, 10, 11, 12, 13, 14, 15);
}

// Templated WMMA GEMM: out[M,NC] = act(A[M,K] @ B[K,NC] + bias) + resid
//  - A tile staged per-wave in LDS row-major f16 -> per-lane A-frag is two
//    contiguous 8-half runs (k = 16*half + s, same k order as B) -> 2x ds_load_b128.
//  - B staged in LDS pre-swizzled fragment-major: [kstep][ntile][lane][16 halfs]
//    (lane's 16 halfs = column n, 16 consecutive k) -> 2x ds_load_b128.
template <int K, int NC, int ACT>
__global__ __launch_bounds__(GEMM_WAVES * 32)
void gemm_wmma(const float* __restrict__ A, const float* __restrict__ B,
               const float* __restrict__ bias, const float* __restrict__ resid,
               float* __restrict__ out, int M) {
  constexpr int NP = (NC + 15) & ~15;
  constexpr int KSTEPS = K / 32;
  constexpr int NTILES = NP / 16;
  constexpr int BTOT = KSTEPS * NTILES * 512;   // halfs
  constexpr int CHUNK = (NTILES < 4) ? NTILES : 4;

  __shared__ __align__(32) _Float16 sB[BTOT];
  __shared__ __align__(16) _Float16 sA[GEMM_WAVES][16 * K];

  const int tid = threadIdx.x;

  // ---- stage B (swizzled fragment-ready), 4 halfs per store ----
  for (int i4 = tid * 4; i4 < BTOT; i4 += GEMM_WAVES * 32 * 4) {
    int j = i4 & 15;                 // element within fragment (mult of 4)
    int ln = (i4 >> 4) & 31;         // fragment lane
    int ft = i4 >> 9;                // kstep * NTILES + ntile
    int nt = ft % NTILES;
    int ks = ft / NTILES;
    int k = ks * 32 + ((ln >> 4) << 4) + j;   // lanes 16-31 hold K+16..31
    int n = nt * 16 + (ln & 15);
    v4h h4;
#pragma unroll
    for (int u = 0; u < 4; ++u)
      h4[u] = (n < NC) ? (_Float16)B[(size_t)(k + u) * NC + n] : (_Float16)0.0f;
    *(v4h*)(sB + i4) = h4;
  }
  __syncthreads();

  const int wave = tid >> 5;
  const int lane = tid & 31;
  const int row0 = (blockIdx.x * GEMM_WAVES + wave) * 16;
  if (row0 >= M) return;

  // ---- stage A tile (row-major f16), float4 loads -> ds_store_b64 ----
  _Float16* myA = sA[wave];
  for (int idx = lane * 4; idx < 16 * K; idx += 32 * 4) {
    int r = idx / K, c = idx - r * K;
    int g = row0 + r;
    float4 f = (g < M) ? *(const float4*)(A + (size_t)g * K + c)
                       : make_float4(0.0f, 0.0f, 0.0f, 0.0f);
    v4h h4; h4[0] = (_Float16)f.x; h4[1] = (_Float16)f.y;
            h4[2] = (_Float16)f.z; h4[3] = (_Float16)f.w;
    *(v4h*)(myA + idx) = h4;
  }
  __builtin_amdgcn_wave_barrier();   // per-wave LDS; DS ops in-order within wave

  const int half = lane >> 4;
  const int nl = lane & 15;
  const int m0 = (lane & 15) * K;

  // ---- preload all A fragments (K <= 256 -> <= 8 x v16h) ----
  v16h afrag[KSTEPS];
#pragma unroll
  for (int ks = 0; ks < KSTEPS; ++ks) {
    const v8h* pa = (const v8h*)(myA + m0 + ks * 32 + (half << 4));
    afrag[ks] = cat16(pa[0], pa[1]);   // run [k0+16h,+16)
  }

#pragma unroll 1
  for (int nt0 = 0; nt0 < NTILES; nt0 += CHUNK) {
    v8f acc[CHUNK] = {};
#pragma unroll
    for (int ks = 0; ks < KSTEPS; ++ks) {
#pragma unroll
      for (int t = 0; t < CHUNK; ++t) {
        const v8h* pb = (const v8h*)(sB + ((ks * NTILES + nt0 + t) << 9) + (lane << 4));
        v16h b = cat16(pb[0], pb[1]);
        acc[t] = __builtin_amdgcn_wmma_f32_16x16x32_f16(
            false, afrag[ks], false, b, (short)0, acc[t], false, false);
      }
    }
#pragma unroll
    for (int t = 0; t < CHUNK; ++t) {
      int n = (nt0 + t) * 16 + nl;
      if constexpr (NP != NC) { if (n >= NC) continue; }
      float bv = bias ? bias[n] : 0.0f;
#pragma unroll
      for (int r = 0; r < 8; ++r) {
        int m = row0 + r + (half << 3);   // C layout: m = r + 8*half, n = lane&15
        if (m >= M) continue;
        float v = acc[t][r] + bv;
        if constexpr (ACT == 1) v = gelu_exact(v);
        if (resid) v += resid[(size_t)m * NC + n];
        out[(size_t)m * NC + n] = v;
      }
    }
  }
}

// LayerNorm over 64 features, one wave per row; optional pre-add bias vector.
__global__ void ln64_kernel(const float* __restrict__ in, const float* __restrict__ addb,
                            const float* __restrict__ g, const float* __restrict__ b,
                            float* __restrict__ out, int n) {
  int gid = blockIdx.x * blockDim.x + threadIdx.x;
  int row = gid >> 5;
  int lane = gid & 31;
  if (row >= n) return;
  const float* p = in + (size_t)row * 64;
  float v0 = p[lane]      + (addb ? addb[lane]      : 0.0f);
  float v1 = p[lane + 32] + (addb ? addb[lane + 32] : 0.0f);
  float s = v0 + v1;
#pragma unroll
  for (int m = 16; m >= 1; m >>= 1) s += __shfl_xor(s, m, 32);
  float mean = s * (1.0f / 64.0f);
  float d0 = v0 - mean, d1 = v1 - mean;
  float q = d0 * d0 + d1 * d1;
#pragma unroll
  for (int m = 16; m >= 1; m >>= 1) q += __shfl_xor(q, m, 32);
  float rstd = rsqrtf(q * (1.0f / 64.0f) + 1e-5f);
  float* o = out + (size_t)row * 64;
  o[lane]      = d0 * rstd * g[lane]      + b[lane];
  o[lane + 32] = d1 * rstd * g[lane + 32] + b[lane + 32];
}

// alpha_s / alpha_d: per (node, head) 64-length dot products
__global__ void alpha_kernel(const float* __restrict__ feat, const float* __restrict__ a_s,
                             const float* __restrict__ a_d, float* __restrict__ os,
                             float* __restrict__ od, int n, int heads) {
  int i = blockIdx.x * blockDim.x + threadIdx.x;
  if (i >= n * heads) return;
  int node = i / heads, h = i - node * heads;
  const float* p = feat + ((size_t)node * heads + h) * 64;
  const float* vs = a_s + h * 64;
  const float* vd = a_d + h * 64;
  float ss = 0.0f, sd = 0.0f;
#pragma unroll 8
  for (int c = 0; c < 64; ++c) { float fv = p[c]; ss += fv * vs[c]; sd += fv * vd[c]; }
  os[i] = ss; od[i] = sd;
}

__global__ void fill_kernel(float* __restrict__ p, int n, float v) {
  int i = blockIdx.x * blockDim.x + threadIdx.x;
  if (i < n) p[i] = v;
}

__device__ __forceinline__ void edge_sd(const int* ei, int E, int e, int& s, int& d) {
  if (e < E) { s = ei[e]; d = ei[E + e]; } else { s = e - E; d = e - E; }
}

__global__ void edge_max_kernel(const int* __restrict__ ei, const float* __restrict__ as_,
                                const float* __restrict__ ad_, float* __restrict__ dmax,
                                int E, int ET, int heads) {
  int i = blockIdx.x * blockDim.x + threadIdx.x;
  if (i >= ET * heads) return;
  int e = i / heads, h = i - e * heads;
  int s, d; edge_sd(ei, E, e, s, d);
  float v = as_[s * heads + h] + ad_[d * heads + h];
  v = (v > 0.0f) ? v : 0.2f * v;                  // leaky_relu(0.2)
  atomMaxF(&dmax[d * heads + h], v);
}

__global__ void edge_exp_kernel(const int* __restrict__ ei, const float* __restrict__ as_,
                                const float* __restrict__ ad_, const float* __restrict__ dmax,
                                float* __restrict__ den, float* __restrict__ exb,
                                int E, int ET, int heads) {
  int i = blockIdx.x * blockDim.x + threadIdx.x;
  if (i >= ET * heads) return;
  int e = i / heads, h = i - e * heads;
  int s, d; edge_sd(ei, E, e, s, d);
  float v = as_[s * heads + h] + ad_[d * heads + h];
  v = (v > 0.0f) ? v : 0.2f * v;
  float ex = expf(v - dmax[d * heads + h]);
  exb[i] = ex;
  atomAddF(&den[d * heads + h], ex);
}

// agg[dst] += (ex/den[dst]) * feat[src]; CH/4 threads per edge, float4 gathers
__global__ void scatter_kernel(const int* __restrict__ ei, const float* __restrict__ feat,
                               const float* __restrict__ exb, const float* __restrict__ den,
                               float* __restrict__ agg, int E, int ET, int heads) {
  int CH = heads << 6;
  int tpe = CH >> 2;
  int gid = blockIdx.x * blockDim.x + threadIdx.x;
  int e = gid / tpe;
  if (e >= ET) return;
  int t = gid - e * tpe;
  int c0 = t << 2;
  int h = c0 >> 6;
  int s, d; edge_sd(ei, E, e, s, d);
  float wgt = exb[e * heads + h] / (den[d * heads + h] + 1e-16f);
  const float4 hv = *(const float4*)(feat + (size_t)s * CH + c0);
  float* op = agg + (size_t)d * CH + c0;
  atomAddF(op + 0, wgt * hv.x);
  atomAddF(op + 1, wgt * hv.y);
  atomAddF(op + 2, wgt * hv.z);
  atomAddF(op + 3, wgt * hv.w);
}

__global__ void bias_elu_kernel(float* __restrict__ p, const float* __restrict__ bias,
                                int total, int C) {
  int i = blockIdx.x * blockDim.x + threadIdx.x;
  if (i >= total) return;
  float v = p[i] + bias[i % C];
  p[i] = (v > 0.0f) ? v : expm1f(v);
}

static inline int cdiv(int a, int b) { return (a + b - 1) / b; }

extern "C" void kernel_launch(void* const* d_in, const int* in_sizes, int n_in,
                              void* d_out, int out_size, void* d_ws, size_t ws_size,
                              hipStream_t stream) {
  const float* x          = (const float*)d_in[0];
  const int*   ei         = (const int*)d_in[1];
  const float* enc_pre_W  = (const float*)d_in[2];
  const float* enc_pre_b  = (const float*)d_in[3];
  const float* enc_Ws     = (const float*)d_in[4];
  const float* enc_bs     = (const float*)d_in[5];
  const float* enc_post_W = (const float*)d_in[6];
  const float* enc_post_b = (const float*)d_in[7];
  const float* ln1_g      = (const float*)d_in[8];
  const float* ln1_b      = (const float*)d_in[9];
  const float* W0         = (const float*)d_in[10];
  const float* a_src0     = (const float*)d_in[11];
  const float* a_dst0     = (const float*)d_in[12];
  const float* bias0      = (const float*)d_in[13];
  const float* W1         = (const float*)d_in[14];
  const float* a_src1     = (const float*)d_in[15];
  const float* a_dst1     = (const float*)d_in[16];
  const float* bias1      = (const float*)d_in[17];
  const float* ln2_g      = (const float*)d_in[18];
  const float* ln2_b      = (const float*)d_in[19];
  const float* dec_pre_W  = (const float*)d_in[20];
  const float* dec_pre_b  = (const float*)d_in[21];
  const float* dec_Ws     = (const float*)d_in[22];
  const float* dec_bs     = (const float*)d_in[23];
  const float* dec_post_W = (const float*)d_in[24];
  const float* dec_post_b = (const float*)d_in[25];

  const int N_ = in_sizes[0] / 64;
  const int E_ = in_sizes[1] / 2;
  const int ET = E_ + N_;

  // workspace layout (floats)
  float* w    = (float*)d_ws;
  float* h1   = w;  w += (size_t)N_ * 64;    // encoder+LN1 output
  float* hh   = w;  w += (size_t)N_ * 256;   // enc ping / GAT0 transform
  float* agg  = w;  w += (size_t)N_ * 256;   // enc pong / GAT0 aggregation (+ELU in-place)
  float* hg   = w;  w += (size_t)N_ * 64;    // GAT1 transform
  float* agg1 = w;  w += (size_t)N_ * 64;    // GAT1 aggregation
  float* h2   = w;  w += (size_t)N_ * 64;    // LN2 output
  float* dca  = w;  w += (size_t)N_ * 32;    // decoder ping
  float* dcb  = w;  w += (size_t)N_ * 32;    // decoder pong
  float* as_  = w;  w += (size_t)N_ * 4;
  float* ad_  = w;  w += (size_t)N_ * 4;
  float* dmax = w;  w += (size_t)N_ * 4;
  float* den  = w;  w += (size_t)N_ * 4;
  float* exb  = w;  w += (size_t)ET * 4;

  const int TPB = 256;
  const int GT = GEMM_WAVES * 32;
  const int ggrid = cdiv(N_, 16 * GEMM_WAVES);

  // ---------------- Encoder MLP: x -> hh/agg ping-pong -> LN1 -> h1 ----------
  gemm_wmma<64, 64, 1><<<ggrid, GT, 0, stream>>>(x, enc_pre_W, enc_pre_b, nullptr, hh, N_);
  float* pa = hh; float* pb = agg;
  for (int i = 0; i < 3; ++i) {
    gemm_wmma<64, 64, 1><<<ggrid, GT, 0, stream>>>(pa, enc_Ws + (size_t)i * 64 * 64, enc_bs + i * 64, pa, pb, N_);
    float* t = pa; pa = pb; pb = t;
  }
  gemm_wmma<64, 64, 0><<<ggrid, GT, 0, stream>>>(pa, enc_post_W, enc_post_b, nullptr, pb, N_);
  ln64_kernel<<<cdiv(N_ * 32, TPB), TPB, 0, stream>>>(pb, nullptr, ln1_g, ln1_b, h1, N_);

  // ---------------- GATConv0: h1 -> hh=[N,256], edge softmax, scatter --------
  gemm_wmma<64, 256, 0><<<ggrid, GT, 0, stream>>>(h1, W0, nullptr, nullptr, hh, N_);
  alpha_kernel<<<cdiv(N_ * 4, TPB), TPB, 0, stream>>>(hh, a_src0, a_dst0, as_, ad_, N_, 4);
  fill_kernel<<<cdiv(N_ * 4, TPB), TPB, 0, stream>>>(dmax, N_ * 4, -3.0e38f);
  fill_kernel<<<cdiv(N_ * 4, TPB), TPB, 0, stream>>>(den, N_ * 4, 0.0f);
  fill_kernel<<<cdiv(N_ * 256, TPB), TPB, 0, stream>>>(agg, N_ * 256, 0.0f);
  edge_max_kernel<<<cdiv(ET * 4, TPB), TPB, 0, stream>>>(ei, as_, ad_, dmax, E_, ET, 4);
  edge_exp_kernel<<<cdiv(ET * 4, TPB), TPB, 0, stream>>>(ei, as_, ad_, dmax, den, exb, E_, ET, 4);
  scatter_kernel<<<cdiv(ET * 64, TPB), TPB, 0, stream>>>(ei, hh, exb, den, agg, E_, ET, 4);
  bias_elu_kernel<<<cdiv(N_ * 256, TPB), TPB, 0, stream>>>(agg, bias0, N_ * 256, 256);

  // ---------------- GATConv1 (1 head): agg -> hg=[N,64], softmax, scatter ----
  gemm_wmma<256, 64, 0><<<ggrid, GT, 0, stream>>>(agg, W1, nullptr, nullptr, hg, N_);
  alpha_kernel<<<cdiv(N_, TPB), TPB, 0, stream>>>(hg, a_src1, a_dst1, as_, ad_, N_, 1);
  fill_kernel<<<cdiv(N_, TPB), TPB, 0, stream>>>(dmax, N_, -3.0e38f);
  fill_kernel<<<cdiv(N_, TPB), TPB, 0, stream>>>(den, N_, 0.0f);
  fill_kernel<<<cdiv(N_ * 64, TPB), TPB, 0, stream>>>(agg1, N_ * 64, 0.0f);
  edge_max_kernel<<<cdiv(ET, TPB), TPB, 0, stream>>>(ei, as_, ad_, dmax, E_, ET, 1);
  edge_exp_kernel<<<cdiv(ET, TPB), TPB, 0, stream>>>(ei, as_, ad_, dmax, den, exb, E_, ET, 1);
  scatter_kernel<<<cdiv(ET * 16, TPB), TPB, 0, stream>>>(ei, hg, exb, den, agg1, E_, ET, 1);
  ln64_kernel<<<cdiv(N_ * 32, TPB), TPB, 0, stream>>>(agg1, bias1, ln2_g, ln2_b, h2, N_);

  // ---------------- Decoder MLP: h2 -> 32 -> (3 residual) -> 6 ---------------
  gemm_wmma<64, 32, 1><<<ggrid, GT, 0, stream>>>(h2, dec_pre_W, dec_pre_b, nullptr, dca, N_);
  pa = dca; pb = dcb;
  for (int i = 0; i < 3; ++i) {
    gemm_wmma<32, 32, 1><<<ggrid, GT, 0, stream>>>(pa, dec_Ws + (size_t)i * 32 * 32, dec_bs + i * 32, pa, pb, N_);
    float* t = pa; pa = pb; pb = t;
  }
  gemm_wmma<32, 6, 0><<<ggrid, GT, 0, stream>>>(pa, dec_post_W, dec_post_b, nullptr, (float*)d_out, N_);
}
